// MultiHead_3805341024435
// MI455X (gfx1250) — matrix operand order repaired and checked by reading, not compiled, not run on hardware
//
#include <hip/hip_runtime.h>

#define HEADS 16
#define GROUPS 4
#define CIN 1024
#define COUT 1024
#define BATCH 4
#define SEQ 1024
#define DH 64
#define KG 256            // CIN / GROUPS

#define QOFF  ((size_t)0)
#define KOFF  ((size_t)(BATCH*HEADS*SEQ*DH))          // 4M elements
#define VOFF  ((size_t)(2*BATCH*HEADS*SEQ*DH))        // 8M elements

// 1/sqrt(64) * log2(e): puts attention scores in the exp2 domain so the
// softmax uses the native base-2 v_exp_f32 with no per-call multiply.
#define QSCALE (0.125f * 1.44269504088896340736f)

typedef __attribute__((ext_vector_type(16))) __bf16    v16bf;
typedef __attribute__((ext_vector_type(8)))  __bf16    v8bf;
typedef __attribute__((ext_vector_type(2)))  __bf16    v2bf;
typedef __attribute__((ext_vector_type(8)))  float     v8f;
typedef __attribute__((ext_vector_type(8)))  unsigned  v8u;
typedef __attribute__((ext_vector_type(4)))  unsigned  v4u;

// Pack two f32 into two bf16 with one hardware instruction.
static __device__ __forceinline__ unsigned pk2(float lo, float hi) {
    unsigned r;
    asm("v_cvt_pk_bf16_f32 %0, %1, %2" : "=v"(r) : "v"(lo), "v"(hi));
    return r;
}
static __device__ __forceinline__ v2bf pk2v(float lo, float hi) {
    return __builtin_bit_cast(v2bf, pk2(lo, hi));
}

static __device__ __forceinline__ v16bf cat8(v8bf lo, v8bf hi) {
    v16bf r;
#pragma unroll
    for (int i = 0; i < 8; ++i) { r[i] = lo[i]; r[i + 8] = hi[i]; }
    return r;
}

static __device__ __forceinline__ v8f wmma_bf16(v16bf a, v16bf b, v8f c) {
    return __builtin_amdgcn_wmma_f32_16x16x32_bf16(false, a, false, b, (short)0, c, false, false);
}

// ---------------------------------------------------------------------------
// Stage 1: grouped QKV projection.  Per block: 64 rows x 64 out-channels
// (each wave: 16 rows x 64 channels -> one A fragment feeds 4 WMMAs).
//   Q (p=0): ws[QOFF] layout [b][h][s][d], scaled by QSCALE
//   K (p=1): ws[KOFF] layout [b][h][s][d]
//   V (p=2): ws[VOFF] layout [b][h][d][s]   (transposed, direct D-layout store)
// ---------------------------------------------------------------------------
__global__ __launch_bounds__(128) void qkv_proj_kernel(
    const float* __restrict__ xq, const float* __restrict__ xk, const float* __restrict__ xv,
    const float* __restrict__ wq, const float* __restrict__ bq,
    const float* __restrict__ wk, const float* __restrict__ bk,
    const float* __restrict__ wv, const float* __restrict__ bv,
    __bf16* __restrict__ ws)
{
    const int p  = blockIdx.z;              // 0=q 1=k 2=v
    const int n0 = blockIdx.y * 64;         // first output channel of tile
    const int g  = n0 >> 8;                 // group (256 out-ch per group)
    const int m0 = blockIdx.x * 64;         // first row of tile
    const int t  = threadIdx.x;
    const int w  = t >> 5;                  // wave id 0..3
    const int l  = t & 31;
    const int lh = l & 15;
    const int hi = l >> 4;                  // lane half

    const float* x    = (p == 0) ? xq : (p == 1) ? xk : xv;
    const float* wm   = (p == 0) ? wq : (p == 1) ? wk : wv;
    const float* bias = (p == 0) ? bq : (p == 1) ? bk : bv;

    __shared__ __bf16 Wb[64][KG];           // Wb[n][k] = W[g][k][n0+n]  (32 KB)
    __shared__ float  bs[64];
    __shared__ float  tb[4][16][16];        // per-wave transpose buffer (4 KB)

    // Stage weights into LDS transposed; coalesced float2 + packed convert.
    for (int it = 0; it < 64; ++it) {
        int lin = it * 128 + t;             // 8192 float2 pairs
        int n2 = (lin & 31) * 2, k = lin >> 5;
        float2 f = *(const float2*)(wm + (size_t)(g * KG + k) * 256 + n0 + n2);
        v2bf pv = pk2v(f.x, f.y);
        Wb[n2][k]     = pv[0];
        Wb[n2 + 1][k] = pv[1];
    }
    if (t < 64) bs[t] = bias[n0 + t];
    __syncthreads();

    const int row = m0 + w * 16 + lh;       // this lane's A-matrix row
    const int b   = row >> 10;
    const float* xrow = x + (size_t)row * CIN + g * KG;

    v8f acc[4] = {{}, {}, {}, {}};
#pragma unroll
    for (int k0 = 0; k0 < KG; k0 += 32) {
        // A fragment: lanes<16 -> K {0..15}; lanes>=16 -> K {16..31}
        const int o1 = k0 + (hi ? 16 : 0);
        const float4* pa = (const float4*)(xrow + o1);
        float4 f0 = pa[0], f1 = pa[1];
        const float4* pb = (const float4*)(xrow + o1 + 8);
        float4 f2 = pb[0], f3 = pb[1];
        v8u ua;
        ua[0] = pk2(f0.x, f0.y); ua[1] = pk2(f0.z, f0.w);
        ua[2] = pk2(f1.x, f1.y); ua[3] = pk2(f1.z, f1.w);
        ua[4] = pk2(f2.x, f2.y); ua[5] = pk2(f2.z, f2.w);
        ua[6] = pk2(f3.x, f3.y); ua[7] = pk2(f3.z, f3.w);
        v16bf a = __builtin_bit_cast(v16bf, ua);
        const int kb = k0 + (hi ? 16 : 0);
#pragma unroll
        for (int j = 0; j < 4; ++j)
            acc[j] = wmma_bf16(a, *(const v16bf*)&Wb[j * 16 + lh][kb], acc[j]);
    }

    const int   swave = (m0 + w * 16) & 1023;
    const float scale = (p == 0) ? QSCALE : 1.0f;

#pragma unroll
    for (int j = 0; j < 4; ++j) {
        const int nt = n0 + j * 16;
        const float bn = bs[j * 16 + lh];
        v8f a = acc[j];
#pragma unroll
        for (int r = 0; r < 8; ++r) a[r] = (a[r] + bn) * scale;

        if (p == 2) {
            // V: lane owns channel nt+lh, rows s contiguous -> direct b128 store
            const int ngl = nt + lh, hh = ngl >> 6, dd = ngl & 63;
            v4u st;
            st[0] = pk2(a[0], a[1]); st[1] = pk2(a[2], a[3]);
            st[2] = pk2(a[4], a[5]); st[3] = pk2(a[6], a[7]);
            *(v4u*)(ws + VOFF + ((size_t)(b * HEADS + hh) * DH + dd) * SEQ
                    + swave + hi * 8) = st;
        } else {
            // Q/K: transpose 16x16 tile through wave-private LDS buffer
#pragma unroll
            for (int r = 0; r < 8; ++r) tb[w][r + hi * 8][lh] = a[r];
            asm volatile("s_wait_dscnt 0" ::: "memory");
            const int c0 = hi * 8;
            v4u st;
            st[0] = pk2(tb[w][lh][c0 + 0], tb[w][lh][c0 + 1]);
            st[1] = pk2(tb[w][lh][c0 + 2], tb[w][lh][c0 + 3]);
            st[2] = pk2(tb[w][lh][c0 + 4], tb[w][lh][c0 + 5]);
            st[3] = pk2(tb[w][lh][c0 + 6], tb[w][lh][c0 + 7]);
            const int nch = nt + c0, hh = nch >> 6, dd = nch & 63;
            *(v4u*)(ws + ((p == 0) ? QOFF : KOFF)
                    + ((size_t)(b * HEADS + hh) * SEQ + swave + lh) * DH + dd) = st;
            asm volatile("s_wait_dscnt 0" ::: "memory");
        }
    }
}

// ---------------------------------------------------------------------------
// Stage 2: flash attention with restricted softmax, K-step = 64.
// One wave per 16-row q-tile; online softmax init m=0, l=1 reproduces
// max(rowmax,0) clamping and the exp(-smax) margin term; all exponentials
// are native base-2 (scores pre-scaled by log2e in stage 1).
// ---------------------------------------------------------------------------
__global__ __launch_bounds__(128) void attn_kernel(
    const __bf16* __restrict__ ws, float* __restrict__ out)
{
    const int b = blockIdx.z, h = blockIdx.y;
    const int t = threadIdx.x;
    const int w = t >> 5;
    const int l = t & 31;
    const int lh = l & 15;
    const int hi = l >> 4;
    const int q0 = (blockIdx.x * 4 + w) * 16;

    const __bf16* Qw = ws + QOFF + (size_t)(b * HEADS + h) * SEQ * DH;  // [s][d]
    const __bf16* Kw = ws + KOFF + (size_t)(b * HEADS + h) * SEQ * DH;  // [s][d]
    const __bf16* Vw = ws + VOFF + (size_t)(b * HEADS + h) * DH * SEQ;  // [d][s]

    __shared__ __bf16 Pl[4][16][64];   // per-wave P-tile transpose buffer (8 KB)

    // Q fragments (two K=32 chunks covering d=64), kept in registers
    v16bf aq0, aq1;
    {
        const __bf16* qr = Qw + (size_t)(q0 + lh) * DH;
        const int o = hi ? 16 : 0;
        aq0 = cat8(*(const v8bf*)(qr + o),      *(const v8bf*)(qr + o + 8));
        aq1 = cat8(*(const v8bf*)(qr + 32 + o), *(const v8bf*)(qr + 40 + o));
    }

    v8f oo[4] = {{}, {}, {}, {}};
    float mr[8], lr[8];
#pragma unroll
    for (int r = 0; r < 8; ++r) { mr[r] = 0.0f; lr[r] = 1.0f; }

    for (int kt = 0; kt < SEQ; kt += 64) {
        // ---- score tiles: 16x64 of scores = 8 WMMAs
        v8f sc[4];
#pragma unroll
        for (int j = 0; j < 4; ++j) {
            const __bf16* kr = Kw + (size_t)(kt + j * 16 + lh) * DH + (hi ? 16 : 0);
            v8f s = {};
            s = wmma_bf16(aq0, *(const v16bf*)kr,        s);
            s = wmma_bf16(aq1, *(const v16bf*)(kr + 32), s);
            sc[j] = s;
        }

        // ---- online restricted softmax over the 64 new columns (base 2)
#pragma unroll
        for (int r = 0; r < 8; ++r) {
            float v = fmaxf(fmaxf(sc[0][r], sc[1][r]), fmaxf(sc[2][r], sc[3][r]));
            v = fmaxf(v, __shfl_xor(v, 1, 32));
            v = fmaxf(v, __shfl_xor(v, 2, 32));
            v = fmaxf(v, __shfl_xor(v, 4, 32));
            v = fmaxf(v, __shfl_xor(v, 8, 32));
            const float mn = fmaxf(mr[r], v);
            const float al = __builtin_amdgcn_exp2f(mr[r] - mn);
            const float p0 = __builtin_amdgcn_exp2f(sc[0][r] - mn);
            const float p1 = __builtin_amdgcn_exp2f(sc[1][r] - mn);
            const float p2 = __builtin_amdgcn_exp2f(sc[2][r] - mn);
            const float p3 = __builtin_amdgcn_exp2f(sc[3][r] - mn);
            float rs = (p0 + p1) + (p2 + p3);
            rs += __shfl_xor(rs, 1, 32);
            rs += __shfl_xor(rs, 2, 32);
            rs += __shfl_xor(rs, 4, 32);
            rs += __shfl_xor(rs, 8, 32);
            lr[r] = lr[r] * al + rs;
            mr[r] = mn;
            const int m = r + hi * 8;
            v2bf pp01 = pk2v(p0, p1);
            v2bf pp23 = pk2v(p2, p3);
            Pl[w][m][lh]      = pp01[0];
            Pl[w][m][lh + 16] = pp01[1];
            Pl[w][m][lh + 32] = pp23[0];
            Pl[w][m][lh + 48] = pp23[1];
            oo[0][r] *= al; oo[1][r] *= al; oo[2][r] *= al; oo[3][r] *= al;
        }
        asm volatile("s_wait_dscnt 0" ::: "memory");

        // ---- reload P in A-matrix layout (D->A transpose via LDS)
        const int po = hi ? 16 : 0;
        v16bf ap0 = cat8(*(const v8bf*)&Pl[w][lh][po],
                         *(const v8bf*)&Pl[w][lh][8 + po]);
        v16bf ap1 = cat8(*(const v8bf*)&Pl[w][lh][32 + po],
                         *(const v8bf*)&Pl[w][lh][40 + po]);

        // ---- O += P x V  (8 WMMAs covering d=64, k=64)
#pragma unroll
        for (int c2 = 0; c2 < 4; ++c2) {
            const __bf16* vb = Vw + (size_t)(c2 * 16 + lh) * SEQ + kt + (hi ? 16 : 0);
            oo[c2] = wmma_bf16(ap0, *(const v16bf*)vb,        oo[c2]);
            oo[c2] = wmma_bf16(ap1, *(const v16bf*)(vb + 32), oo[c2]);
        }
    }

    // ---- finalize: divide by denominator, store [b][s][h*64+d]
    const int sbase = q0 + hi * 8;
#pragma unroll
    for (int r = 0; r < 8; ++r) {
        const float inv = 1.0f / lr[r];
        float* op = out + ((size_t)b * SEQ + (sbase + r)) * COUT + h * DH;
        op[ 0 + lh] = oo[0][r] * inv;
        op[16 + lh] = oo[1][r] * inv;
        op[32 + lh] = oo[2][r] * inv;
        op[48 + lh] = oo[3][r] * inv;
    }
}

extern "C" void kernel_launch(void* const* d_in, const int* in_sizes, int n_in,
                              void* d_out, int out_size, void* d_ws, size_t ws_size,
                              hipStream_t stream) {
    const float* xq = (const float*)d_in[0];
    const float* xk = (const float*)d_in[1];
    const float* xv = (const float*)d_in[2];
    const float* wq = (const float*)d_in[3];
    const float* bq = (const float*)d_in[4];
    const float* wk = (const float*)d_in[5];
    const float* bk = (const float*)d_in[6];
    const float* wv = (const float*)d_in[7];
    const float* bv = (const float*)d_in[8];
    __bf16* ws = (__bf16*)d_ws;
    float* out = (float*)d_out;
    (void)in_sizes; (void)n_in; (void)out_size; (void)ws_size;

    dim3 g1(BATCH * SEQ / 64, COUT / 64, 3), b1(128);
    hipLaunchKernelGGL(qkv_proj_kernel, g1, b1, 0, stream,
                       xq, xk, xv, wq, bq, wk, bk, wv, bv, ws);

    dim3 g2(SEQ / 64, HEADS, BATCH), b2(128);
    hipLaunchKernelGGL(attn_kernel, g2, b2, 0, stream, (const __bf16*)ws, out);
}
